// Attention_88682484728470
// MI455X (gfx1250) — matrix operand order repaired and checked by reading, not compiled, not run on hardware
//
#include <hip/hip_runtime.h>
#include <cmath>

typedef __bf16 bf16_t;
typedef __attribute__((ext_vector_type(16))) __bf16 v16bf;
typedef __attribute__((ext_vector_type(8)))  __bf16 v8bf;
typedef __attribute__((ext_vector_type(8)))  float  v8f;
typedef __attribute__((ext_vector_type(4)))  float  v4f;

#define DHEAD    64
#define KVCHUNK  64                 // kv rows per staged chunk (4 x 16 tiles)
#define WAVES    8
#define QROWS    (WAVES * 16)       // q rows per block = 128
// fold 1/sqrt(64) and log2(e) into the Q->bf16 conversion (prologue)
#define SCALE_LOG2 (0.125f * 1.4426950408889634f)
#define NEG_BIG   (-1.0e30f)

__device__ __forceinline__ float tmax8(const v8f v) {
  const float a = fmaxf(v[0], v[1]), b = fmaxf(v[2], v[3]);
  const float c = fmaxf(v[4], v[5]), d = fmaxf(v[6], v[7]);
  return fmaxf(fmaxf(a, b), fmaxf(c, d));
}

// Flash attention fwd, causal, fp32 in/out, bf16 WMMA compute.
// Scores computed TRANSPOSED (S^T = K * Q^T): the WMMA C-layout of S^T is
// exactly the A-layout of P for the P*V WMMA, so P never visits LDS and each
// lane owns one softmax row (single shfl_xor(16) per reduction).
__global__ __launch_bounds__(256) void fa_fwd_bf16wmma(
    const float* __restrict__ Kg, const float* __restrict__ Vg,
    const float* __restrict__ Qg, float* __restrict__ Og, int S) {
  // double-buffered K (row-major [kv][d]) and V (transposed [d][kv]) in bf16
  __shared__ bf16_t lds_k[2][KVCHUNK * DHEAD] __attribute__((aligned(32)));
  __shared__ bf16_t lds_v[2][DHEAD * KVCHUNK] __attribute__((aligned(32)));

  const int tid  = threadIdx.x;
  const int wave = tid >> 5;
  const int lane = tid & 31;
  const int half = lane >> 4;          // 0: lanes 0-15, 1: lanes 16-31
  const int ln   = lane & 15;
  const int qblk = blockIdx.x;
  const int bh   = blockIdx.y;
  const int q0w  = qblk * QROWS + wave * 16;  // wave's first q row
  const int q    = q0w + ln;                  // this lane's softmax row
  const size_t base = (size_t)bh * (size_t)S * DHEAD;

  // ---- Q tile as B-layout registers, pre-scaled by 0.125*log2(e) ----
  // B 32x16 bf16: lane holds j<8 -> d = 32*dc + 8*half + j, j>=8 -> d = 32*dc + 16 + 8*half + (j-8)
  const float* qrow = Qg + base + (size_t)q * DHEAD;
  v16bf bq[2];
#pragma unroll
  for (int dc = 0; dc < 2; ++dc)
#pragma unroll
    for (int g = 0; g < 4; ++g) {
      const v4f f = *(const v4f*)(qrow + 32 * dc + 8 * half + 4 * g + 8 * (g >> 1));
#pragma unroll
      for (int e = 0; e < 4; ++e)
        bq[dc][4 * g + e] = (bf16_t)(f[e] * SCALE_LOG2);
    }

  // out accumulator, C layout: lane = d col (16t+ln), VGPR r = row r + 8*half
  v8f acc[4] = {{}, {}, {}, {}};
  float m = NEG_BIG, l = 0.f;   // per-lane row stats (row q, mirrored in lane^16)

  // staging split: thread owns a PAIR of kv rows x 8 d cols (4096 elems total)
  // -> K: 2x ds_store_b128; V-transpose: 8x cvt_pk + 8x ds_store_b32
  const int rp   = tid >> 3;          // row pair 0..31 -> rows 2rp, 2rp+1
  const int cg   = (tid & 7) * 8;     // col group 0..56

  const int kv_end = qblk * QROWS + QROWS;   // causal bound for the block

  // prefetch chunk 0 into registers
  float k0[8], k1[8], v0[8], v1[8];
  {
    const float* kp = Kg + base + (size_t)(2 * rp) * DHEAD + cg;
    const float* vp = Vg + base + (size_t)(2 * rp) * DHEAD + cg;
#pragma unroll
    for (int j = 0; j < 8; ++j) {
      k0[j] = kp[j]; k1[j] = kp[DHEAD + j];
      v0[j] = vp[j]; v1[j] = vp[DHEAD + j];
    }
  }

  int buf = 0;
  for (int kv = 0; kv < kv_end; kv += KVCHUNK, buf ^= 1) {
    // ---- commit staged registers to LDS ----
    {
      v8bf ka, kb;
#pragma unroll
      for (int j = 0; j < 8; ++j) { ka[j] = (bf16_t)k0[j]; kb[j] = (bf16_t)k1[j]; }
      *(v8bf*)&lds_k[buf][(2 * rp) * DHEAD + cg]     = ka;
      *(v8bf*)&lds_k[buf][(2 * rp + 1) * DHEAD + cg] = kb;
#pragma unroll
      for (int j = 0; j < 8; ++j) {
        // adjacent kv pair at fixed d: one packed 32-bit store
        v8bf pk;
        pk[0] = (bf16_t)v0[j];
        pk[1] = (bf16_t)v1[j];
        *(short2*)&lds_v[buf][(cg + j) * KVCHUNK + 2 * rp] = *(short2*)&pk;
      }
    }
    __syncthreads();

    // ---- issue next chunk's global loads (overlap with compute) ----
    if (kv + KVCHUNK < kv_end) {
      const float* kp = Kg + base + (size_t)(kv + KVCHUNK + 2 * rp) * DHEAD + cg;
      const float* vp = Vg + base + (size_t)(kv + KVCHUNK + 2 * rp) * DHEAD + cg;
#pragma unroll
      for (int j = 0; j < 8; ++j) {
        k0[j] = kp[j]; k1[j] = kp[DHEAD + j];
        v0[j] = vp[j]; v1[j] = vp[DHEAD + j];
      }
    }

    // wave-uniform causal skip: chunk entirely above the diagonal for this wave
    if (kv <= q0w + 15) {
      // ---- S^T tiles: A = K tile (M=kv, K=d), B = Q regs; C lane = q ----
      // dc OUTER, tile INNER: consecutive WMMAs hit different accumulators,
      // so the bf16 WMMA->WMMA RAW hazard slots fill with real work (no nops)
      v8f s[4] = {{}, {}, {}, {}};
#pragma unroll
      for (int dc = 0; dc < 2; ++dc)
#pragma unroll
        for (int t = 0; t < 4; ++t) {
          // A 16x32: lane kv row = 16t+ln; elems j<8 -> d=32dc+8half+j,
          // elems j>=8 -> d=32dc+16+8half+(j-8)
          const bf16_t* kb =
              &lds_k[buf][(16 * t + ln) * DHEAD + 32 * dc + 8 * half];
          const v8bf lo = *(const v8bf*)kb;
          const v8bf hi = *(const v8bf*)(kb + 16);
          const v16bf ak = __builtin_shufflevector(
              lo, hi, 0, 1, 2, 3, 4, 5, 6, 7, 8, 9, 10, 11, 12, 13, 14, 15);
          s[t] = __builtin_amdgcn_wmma_f32_16x16x32_bf16(
              false, ak, false, bq[dc], (short)0, s[t], false, false);
        }

      // ---- causal mask, only on the diagonal chunk (wave-uniform) ----
      if (kv + KVCHUNK - 1 > q0w) {
#pragma unroll
        for (int t = 0; t < 4; ++t)
#pragma unroll
          for (int r = 0; r < 8; ++r)
            if (kv + 16 * t + 8 * half + r > q) s[t][r] = NEG_BIG;
      }

      // ---- online softmax: tree max + ONE cross-lane exchange ----
      float mc = fmaxf(fmaxf(tmax8(s[0]), tmax8(s[1])),
                       fmaxf(tmax8(s[2]), tmax8(s[3])));
      mc = fmaxf(mc, __shfl_xor(mc, 16, 32));
      const float mn   = fmaxf(m, mc);
      const bool  grew = mn > m;                    // alpha < 1 iff max grew
      // args are <= 0: raw v_exp_f32 (underflow->0) is exactly what we want;
      // avoids the 5-op libm range-fixup sequence per exp
      const float alpha = __builtin_amdgcn_exp2f(m - mn);
      m = mn;

      // P straight into A-layout registers (C layout of S^T == A layout);
      // ap[g] covers kv group 32g..32g+31; per-tile partial sums (tree)
      v16bf ap[2];
      float ts[4];
#pragma unroll
      for (int t = 0; t < 4; ++t) {
        float pv[8];
#pragma unroll
        for (int r = 0; r < 8; ++r) {
          pv[r] = __builtin_amdgcn_exp2f(s[t][r] - mn);
          ap[t >> 1][8 * (t & 1) + r] = (bf16_t)pv[r];
        }
        ts[t] = ((pv[0] + pv[1]) + (pv[2] + pv[3])) +
                ((pv[4] + pv[5]) + (pv[6] + pv[7]));
      }
      l = l * alpha + ((ts[0] + ts[1]) + (ts[2] + ts[3]));

      // rescale accumulator only if some lane's max grew (exact skip: alpha==1)
      if (__ballot(grew)) {
#pragma unroll
        for (int r = 0; r < 8; ++r) {
          const float ar = __shfl(alpha, r + 8 * half, 32);
#pragma unroll
          for (int t = 0; t < 4; ++t) acc[t][r] *= ar;
        }
      }

      // ---- PV: A = P (regs), B = V tiles; g OUTER so consecutive WMMAs
      // write different acc[t] (no dependent back-to-back WMMAs) ----
#pragma unroll
      for (int g = 0; g < 2; ++g)
#pragma unroll
        for (int t = 0; t < 4; ++t) {
          // B 32x16: lane d col = 16t+ln; holds j<8 -> kv = 32g + 8*half + j, j>=8 -> kv = 32g + 16 + 8*half + (j-8)
          const bf16_t* vb =
              &lds_v[buf][(16 * t + ln) * KVCHUNK + 32 * g + 8 * half];
          const v8bf vlo = *(const v8bf*)vb;
          const v8bf vhi = *(const v8bf*)(vb + 16);
          const v16bf bv = __builtin_shufflevector(
              vlo, vhi, 0, 1, 2, 3, 4, 5, 6, 7, 8, 9, 10, 11, 12, 13, 14, 15);
          acc[t] = __builtin_amdgcn_wmma_f32_16x16x32_bf16(
              false, ap[g], false, bv, (short)0, acc[t], false, false);
        }
    }
    __syncthreads();
  }

  // ---- epilogue: merge lane-pair row sums, normalize, store fp32 ----
  l += __shfl_xor(l, 16, 32);
  const float linv = 1.0f / l;
#pragma unroll
  for (int r = 0; r < 8; ++r) {
    const float ir = __shfl(linv, r + 8 * half, 32);  // 1/l of row r+8*half
    const int qr = q0w + r + 8 * half;
#pragma unroll
    for (int t = 0; t < 4; ++t)
      Og[base + (size_t)qr * DHEAD + 16 * t + ln] = acc[t][r] * ir;
  }
}

extern "C" void kernel_launch(void* const* d_in, const int* in_sizes, int n_in,
                              void* d_out, int out_size, void* d_ws, size_t ws_size,
                              hipStream_t stream) {
  const float* K = (const float*)d_in[0];
  const float* V = (const float*)d_in[1];
  const float* Q = (const float*)d_in[2];
  float* O = (float*)d_out;
  (void)n_in; (void)out_size; (void)d_ws; (void)ws_size;

  const long long msz = in_sizes[3];                 // S*S causal mask
  const int S  = (int)llround(sqrt((double)msz));
  const int BH = (int)((long long)in_sizes[0] / ((long long)S * DHEAD));

  dim3 grid(S / QROWS, BH);
  dim3 block(32 * WAVES);
  fa_fwd_bf16wmma<<<grid, block, 0, stream>>>(K, V, Q, O, S);
}
